// ScaledDotProductAttention_20358144983419
// MI455X (gfx1250) — matrix operand order repaired and checked by reading, not compiled, not run on hardware
//
#include <hip/hip_runtime.h>
#include <stdint.h>

typedef __attribute__((ext_vector_type(16))) __bf16 v16bf;
typedef __attribute__((ext_vector_type(8)))  __bf16 v8bf;
typedef __attribute__((ext_vector_type(8)))  float  v8f;

#define B_    32
#define S_    2048
#define D_    64
#define WAVES 8      // waves per block
#define QT    16     // q rows per wave
#define KT    32     // keys per iteration

// LDS row strides (bf16 elements), 16B-aligned rows, conflict-minimized
#define KSTRIDE 72   // K tile: 32 keys x 64 feats
#define VSTRIDE 40   // V^T tile: 64 feat-rows x 32 keys
#define PSTRIDE 40   // P tile: 16 query-rows x 32 keys

#define LOG2E   1.44269504088896340736f
#define QSCALE  (0.125f * LOG2E)              // 1/sqrt(64) folded with log2e

__device__ __forceinline__ void cvt4s(v16bf& d, int off, float4 a, float s) {
    d[off+0]=(__bf16)(a.x*s); d[off+1]=(__bf16)(a.y*s);
    d[off+2]=(__bf16)(a.z*s); d[off+3]=(__bf16)(a.w*s);
}

__global__ __launch_bounds__(256) void attn_fwd_kernel(
    const float* __restrict__ Q, const float* __restrict__ K,
    const float* __restrict__ V, const uint8_t* __restrict__ M,
    float* __restrict__ O)
{
    __shared__ __bf16 sK[KT * KSTRIDE];          // row-major [key][feat]
    __shared__ __bf16 sV[D_ * VSTRIDE];          // transposed [feat][key]
    __shared__ __bf16 sP[WAVES][QT * PSTRIDE];   // per-wave P [query][key]

    const int tid  = threadIdx.x;
    const int wave = tid >> 5;
    const int lane = tid & 31;
    const int half = lane >> 4;   // 0 or 1
    const int col  = lane & 15;   // query index within tile / A-row index

    const int blocks_per_batch = S_ / (QT * WAVES);   // 16
    const int b   = blockIdx.x / blocks_per_batch;
    const int blk = blockIdx.x % blocks_per_batch;
    const int q0  = (blk * WAVES + wave) * QT;

    // ---- Q^T in B-matrix layout, pre-scaled by 0.125*log2e ----
    // B[kfeat][n=query]: lane n=col, kfeat = c*32 + 16*half + e (16 contiguous)
    v16bf qb0, qb1;
    {
        const float* qrow = Q + ((size_t)b * S_ + q0 + col) * D_;
        const float4* p0 = (const float4*)(qrow + half * 16);
        cvt4s(qb0, 0, p0[0], QSCALE); cvt4s(qb0, 4, p0[1], QSCALE);
        cvt4s(qb0, 8, p0[2], QSCALE); cvt4s(qb0, 12, p0[3], QSCALE);
        const float4* p1 = (const float4*)(qrow + 32 + half * 16);
        cvt4s(qb1, 0, p1[0], QSCALE); cvt4s(qb1, 4, p1[1], QSCALE);
        cvt4s(qb1, 8, p1[2], QSCALE); cvt4s(qb1, 12, p1[3], QSCALE);
    }

    v8f acc[4] = {};              // context^T: 4 chunks of [16 feat x 16 query]
    float m_run = -INFINITY;      // per-lane = per-query running max (base-2)
    float l_run = 0.0f;           // per-lane running sum

    for (int k0 = 0; k0 < S_; k0 += KT) {
        // ---- cooperative stage: K row-major bf16, V transposed bf16 ----
        {
            const int krow = tid >> 3;        // 0..31
            const int f0   = (tid & 7) * 8;   // 0..56
            const float4* kg = (const float4*)(K + ((size_t)b * S_ + k0 + krow) * D_ + f0);
            float4 x0 = kg[0], x1 = kg[1];
            v8bf kk;
            kk[0]=(__bf16)x0.x; kk[1]=(__bf16)x0.y; kk[2]=(__bf16)x0.z; kk[3]=(__bf16)x0.w;
            kk[4]=(__bf16)x1.x; kk[5]=(__bf16)x1.y; kk[6]=(__bf16)x1.z; kk[7]=(__bf16)x1.w;
            *(v8bf*)&sK[krow * KSTRIDE + f0] = kk;

            const float4* vg = (const float4*)(V + ((size_t)b * S_ + k0 + krow) * D_ + f0);
            float4 y0 = vg[0], y1 = vg[1];
            sV[(f0+0)*VSTRIDE + krow] = (__bf16)y0.x;
            sV[(f0+1)*VSTRIDE + krow] = (__bf16)y0.y;
            sV[(f0+2)*VSTRIDE + krow] = (__bf16)y0.z;
            sV[(f0+3)*VSTRIDE + krow] = (__bf16)y0.w;
            sV[(f0+4)*VSTRIDE + krow] = (__bf16)y1.x;
            sV[(f0+5)*VSTRIDE + krow] = (__bf16)y1.y;
            sV[(f0+6)*VSTRIDE + krow] = (__bf16)y1.z;
            sV[(f0+7)*VSTRIDE + krow] = (__bf16)y1.w;
        }
        __syncthreads();

        // ---- S^T tiles: D[key][query] = K(16x32) x Q^T(32x16), 2 k-chunks ----
        v8f s0 = {}, s1 = {};
        #pragma unroll
        for (int t = 0; t < 2; ++t) {
            v8f sc = {};
            #pragma unroll
            for (int c = 0; c < 2; ++c) {
                // A = K rows: lane row m=col -> key t*16+col; feats c*32 + 16*half + e (contiguous)
                const v8bf* a0 = (const v8bf*)&sK[(t*16 + col) * KSTRIDE + c*32 + half*16];
                const v8bf* a1 = (const v8bf*)&sK[(t*16 + col) * KSTRIDE + c*32 + half*16 + 8];
                v8bf lo = a0[0], hi = a1[0];
                v16bf ka;
                #pragma unroll
                for (int i = 0; i < 8; ++i) { ka[i] = lo[i]; ka[8+i] = hi[i]; }
                sc = __builtin_amdgcn_wmma_f32_16x16x32_bf16(
                        false, ka, false, (c ? qb1 : qb0), (short)0, sc, false, false);
            }
            if (t == 0) s0 = sc; else s1 = sc;
        }

        // ---- mask bytes (8 consecutive keys per tile), inverted to {0,1} ----
        const uint8_t* mp = M + ((size_t)b * S_ + q0 + col) * S_ + k0 + half*8;
        uint2 u0 = *(const uint2*)(mp);        // tile0 keys 8h..8h+7
        uint2 u1 = *(const uint2*)(mp + 16);   // tile1 keys 16+8h..
        uint32_t n0[2] = { ~u0.x & 0x01010101u, ~u0.y & 0x01010101u };
        uint32_t n1[2] = { ~u1.x & 0x01010101u, ~u1.y & 0x01010101u };

        // ---- online softmax; max over RAW scores (valid upper bound:
        //      softmax is invariant to the shift, masked p's are zeroed below,
        //      matching the reference's exp(-10000-m) == 0.0f underflow) ----
        float t0 = fmaxf(fmaxf(s0[0], s0[1]), fmaxf(s0[2], s0[3]));
        float t1 = fmaxf(fmaxf(s0[4], s0[5]), fmaxf(s0[6], s0[7]));
        float t2 = fmaxf(fmaxf(s1[0], s1[1]), fmaxf(s1[2], s1[3]));
        float t3 = fmaxf(fmaxf(s1[4], s1[5]), fmaxf(s1[6], s1[7]));
        float mx = fmaxf(fmaxf(t0, t1), fmaxf(t2, t3));
        mx = fmaxf(mx, __shfl_xor(mx, 16));
        float mnew  = fmaxf(m_run, mx);
        float alpha = __builtin_amdgcn_exp2f(m_run - mnew);

        // p = exp2(s - m) * notmask   (notmask via v_cvt_f32_ubyte pattern)
        #pragma unroll
        for (int r = 0; r < 8; ++r) {
            float nba = (float)((n0[r >> 2] >> ((r & 3) * 8)) & 0xffu);
            float nbb = (float)((n1[r >> 2] >> ((r & 3) * 8)) & 0xffu);
            s0[r] = __builtin_amdgcn_exp2f(s0[r] - mnew) * nba;
            s1[r] = __builtin_amdgcn_exp2f(s1[r] - mnew) * nbb;
        }
        float a0 = (s0[0] + s0[1]) + (s0[2] + s0[3]);
        float a1 = (s0[4] + s0[5]) + (s0[6] + s0[7]);
        float a2 = (s1[0] + s1[1]) + (s1[2] + s1[3]);
        float a3 = (s1[4] + s1[5]) + (s1[6] + s1[7]);
        float sum = (a0 + a1) + (a2 + a3);
        sum += __shfl_xor(sum, 16);
        l_run = l_run * alpha + sum;
        m_run = mnew;
        #pragma unroll
        for (int ch = 0; ch < 4; ++ch) acc[ch] *= alpha;

        // ---- P: pack 8 consecutive keys per tile -> 2x ds_write_b128 ----
        __bf16* pb = &sP[wave][0];
        {
            v8bf pk0, pk1;
            #pragma unroll
            for (int r = 0; r < 8; ++r) { pk0[r] = (__bf16)s0[r]; pk1[r] = (__bf16)s1[r]; }
            *(v8bf*)&pb[col * PSTRIDE + half*8]      = pk0;   // keys 8h..8h+7
            *(v8bf*)&pb[col * PSTRIDE + 16 + half*8] = pk1;   // keys 16+8h..
        }
        __builtin_amdgcn_wave_barrier();

        // ---- P in B-layout: lane n=col, k = 16*half + e (contiguous 32B) ----
        v16bf pbm;
        {
            const v8bf* pr = (const v8bf*)&pb[col * PSTRIDE + half*16];
            v8bf lo = pr[0], hi = pr[1];
            #pragma unroll
            for (int i = 0; i < 8; ++i) { pbm[i] = lo[i]; pbm[8+i] = hi[i]; }
        }

        // ---- context^T += V^T(16x32) x P(32x16) per 16-wide feat chunk ----
        #pragma unroll
        for (int ch = 0; ch < 4; ++ch) {
            const v8bf* w0 = (const v8bf*)&sV[(ch*16 + col) * VSTRIDE + half*16];
            const v8bf* w1 = (const v8bf*)&sV[(ch*16 + col) * VSTRIDE + half*16 + 8];
            v8bf lo = w0[0], hi = w1[0];
            v16bf va;
            #pragma unroll
            for (int i = 0; i < 8; ++i) { va[i] = lo[i]; va[8+i] = hi[i]; }
            acc[ch] = __builtin_amdgcn_wmma_f32_16x16x32_bf16(
                          false, va, false, pbm, (short)0, acc[ch], false, false);
        }
        __syncthreads();
    }

    // ---- epilogue: out[q][d] = acc^T / l ; 8 consecutive d per lane/chunk ----
    {
        float inv = 1.0f / l_run;
        float* orow = O + ((size_t)b * S_ + q0 + col) * D_ + half * 8;
        #pragma unroll
        for (int ch = 0; ch < 4; ++ch) {
            float4 o0 = make_float4(acc[ch][0]*inv, acc[ch][1]*inv,
                                    acc[ch][2]*inv, acc[ch][3]*inv);
            float4 o1 = make_float4(acc[ch][4]*inv, acc[ch][5]*inv,
                                    acc[ch][6]*inv, acc[ch][7]*inv);
            *(float4*)(orow + ch*16)     = o0;
            *(float4*)(orow + ch*16 + 4) = o1;
        }
    }
}

extern "C" void kernel_launch(void* const* d_in, const int* in_sizes, int n_in,
                              void* d_out, int out_size, void* d_ws, size_t ws_size,
                              hipStream_t stream) {
    const float*   Q = (const float*)d_in[0];
    const float*   K = (const float*)d_in[1];
    const float*   V = (const float*)d_in[2];
    const uint8_t* M = (const uint8_t*)d_in[3];
    float*         O = (float*)d_out;

    dim3 grid(B_ * (S_ / (QT * WAVES)));   // 512 blocks
    dim3 block(256);                        // 8 waves
    attn_fwd_kernel<<<grid, block, 0, stream>>>(Q, K, V, M, O);
}
